// DynamicArchitectureSelector_76982993814142
// MI455X (gfx1250) — matrix operand order repaired and checked by reading, not compiled, not run on hardware
//
#include <hip/hip_runtime.h>
#include <hip/hip_bf16.h>

// ---------------------------------------------------------------------------
// Problem constants (reference: B=2, T=2048, D_MODEL=1024, SEL_DIM=64,
// CONFIGS = [(4,32),(8,64),(16,128)])
// ---------------------------------------------------------------------------
#define BN 2
#define TN 2048
#define DN 1024

typedef __bf16 v16bf __attribute__((ext_vector_type(16)));
typedef __bf16 v8bf  __attribute__((ext_vector_type(8)));
typedef float  v8f   __attribute__((ext_vector_type(8)));

#define WMMA_BF16(a, b, c) \
    __builtin_amdgcn_wmma_f32_16x16x32_bf16(false, (a), false, (b), (short)0, (c), false, false)

// Load 16 bf16 for the A operand: elements 0..15 contiguous at p
// (contiguous k order, matching the B operand fill).
__device__ inline v16bf load_a16(const __bf16* p) {
    v8bf lo = *(const v8bf*)p;
    v8bf hi = *(const v8bf*)(p + 8);
    return __builtin_shufflevector(lo, hi, 0, 1, 2, 3, 4, 5, 6, 7,
                                           8, 9, 10, 11, 12, 13, 14, 15);
}

// ---------------------------------------------------------------------------
// Per-(batch, feature) mean / unbiased std over the T axis.
// ---------------------------------------------------------------------------
__global__ void stats_kernel(const float* __restrict__ x,
                             float* __restrict__ mean, float* __restrict__ stdv) {
    int col = blockIdx.x;                 // [0, B*D)
    int b = col / DN, dch = col % DN;
    const float* base = x + (size_t)b * TN * DN + dch;
    float s = 0.f, s2 = 0.f;
    for (int t = threadIdx.x; t < TN; t += blockDim.x) {
        float v = base[(size_t)t * DN];
        s += v; s2 += v * v;
    }
    __shared__ float sh[256], sh2[256];
    int tid = threadIdx.x;
    sh[tid] = s; sh2[tid] = s2;
    __syncthreads();
    for (int off = 128; off > 0; off >>= 1) {
        if (tid < off) { sh[tid] += sh[tid + off]; sh2[tid] += sh2[tid + off]; }
        __syncthreads();
    }
    if (tid == 0) {
        float m = sh[0] / (float)TN;
        float var = (sh2[0] - (float)TN * m * m) / (float)(TN - 1);
        mean[col] = m;
        stdv[col] = sqrtf(fmaxf(var, 0.f));
    }
}

// feats = x + mean + std + 0.1*(t/T); also emit bf16 copy of x for projections.
__global__ void feats_kernel(const float* __restrict__ x,
                             const float* __restrict__ mean,
                             const float* __restrict__ stdv,
                             __bf16* __restrict__ xb, __bf16* __restrict__ fb) {
    size_t i = (size_t)blockIdx.x * blockDim.x + threadIdx.x;
    size_t total = (size_t)BN * TN * DN;
    if (i >= total) return;
    int dch = (int)(i % DN);
    size_t bt = i / DN;
    int t = (int)(bt % TN);
    int b = (int)(bt / TN);
    float v = x[i];
    xb[i] = (__bf16)v;
    float f = v + mean[b * DN + dch] + stdv[b * DN + dch] + 0.1f * ((float)t / (float)TN);
    fb[i] = (__bf16)f;
}

__global__ void f32_to_bf16(const float* __restrict__ in, __bf16* __restrict__ out, size_t n) {
    size_t i = (size_t)blockIdx.x * blockDim.x + threadIdx.x;
    if (i < n) out[i] = (__bf16)in[i];
}

// ---------------------------------------------------------------------------
// bf16 WMMA GEMM: out = act(A @ W^T + bias); A:[M,K] bf16, W:[N,K] bf16.
// One wave computes a 16 x (16*NSUB) strip: A fragment loaded once per k-step
// and reused across NSUB accumulators. k-loop unrolled 2x (bounded pressure).
// ---------------------------------------------------------------------------
template <int NSUB>
__global__ void gemm16_bf16(const __bf16* __restrict__ A, const __bf16* __restrict__ W,
                            const float* __restrict__ bias,
                            float* __restrict__ outF, __bf16* __restrict__ outB,
                            int M, int N, int K, int doRelu) {
    int tile = blockIdx.x * (blockDim.x >> 5) + (threadIdx.x >> 5);
    int ntn = N / (16 * NSUB);
    int total = (M >> 4) * ntn;
    if (tile >= total) return;           // wave-uniform exit; EXEC all-1 at WMMA
    int tm = tile / ntn, tn = tile % ntn;
    int lane = threadIdx.x & 31;
    int l = lane & 15, hi = lane >> 4;

    const __bf16* arow = A + (size_t)(tm * 16 + l) * K + hi * 16;
    const __bf16* wrow[NSUB];
#pragma unroll
    for (int j = 0; j < NSUB; ++j)
        wrow[j] = W + (size_t)(tn * 16 * NSUB + j * 16 + l) * K + hi * 16;

    v8f c[NSUB];
#pragma unroll
    for (int j = 0; j < NSUB; ++j) c[j] = (v8f){};

#pragma unroll 2
    for (int k = 0; k < K; k += 32) {
        v16bf a = load_a16(arow + k);
#pragma unroll
        for (int j = 0; j < NSUB; ++j) {
            v16bf b = *(const v16bf*)(wrow[j] + k);  // 16 contiguous bf16 of W[n,:]
            c[j] = WMMA_BF16(a, b, c[j]);
        }
    }

    int mrow = tm * 16 + (hi << 3);
#pragma unroll
    for (int j = 0; j < NSUB; ++j) {
        int n = tn * 16 * NSUB + j * 16 + l;
        float bv = bias ? bias[n] : 0.f;
#pragma unroll
        for (int r = 0; r < 8; ++r) {
            float v = c[j][r] + bv;
            if (doRelu) v = fmaxf(v, 0.f);
            size_t o = (size_t)(mrow + r) * N + n;
            if (outF) outF[o] = v; else outB[o] = (__bf16)v;
        }
    }
}

// ---------------------------------------------------------------------------
// Selector head (N=3) + softmax: one thread per token row.
// ---------------------------------------------------------------------------
__global__ void sel3_softmax(const __bf16* __restrict__ h2, const float* __restrict__ w3,
                             const float* __restrict__ b3, float* __restrict__ probs, int rows) {
    int i = blockIdx.x * blockDim.x + threadIdx.x;
    if (i >= rows) return;
    const __bf16* r = h2 + (size_t)i * 64;
    float lg[3];
#pragma unroll
    for (int c = 0; c < 3; ++c) {
        float s = b3[c];
        const float* wc = w3 + c * 64;
        for (int k = 0; k < 64; ++k) s += (float)r[k] * wc[k];
        lg[c] = s;
    }
    float m = fmaxf(lg[0], fmaxf(lg[1], lg[2]));
    float e0 = __expf(lg[0] - m), e1 = __expf(lg[1] - m), e2 = __expf(lg[2] - m);
    float inv = 1.f / (e0 + e1 + e2);
    probs[i * 3 + 0] = e0 * inv;
    probs[i * 3 + 1] = e1 * inv;
    probs[i * 3 + 2] = e2 * inv;
}

// Small f32 matvec for the head-weight projection w = x @ ww^T (N <= 16).
__global__ void matvec_f32(const float* __restrict__ X, const float* __restrict__ W,
                           float* __restrict__ out, int rows, int N, int K) {
    int idx = blockIdx.x * blockDim.x + threadIdx.x;
    if (idx >= rows * N) return;
    int rr = idx / N, n = idx % N;
    const float* xr = X + (size_t)rr * K;
    const float* wr = W + (size_t)n * K;
    float s = 0.f;
    for (int k = 0; k < K; ++k) s += xr[k] * wr[k];
    out[idx] = s;
}

// ---------------------------------------------------------------------------
// Per-config accumulation helper for the score kernel. Template params give
// compile-time H/Dm (immediate-offset addressing) and, critically, the Q/K/W
// pointers arrive directly from kernel arguments (no runtime-indexed pointer
// array), so the backend keeps global address-space inference and emits
// global_load_* (LOADcnt only) instead of flat_load_* (LOADcnt+DScnt).
// Head loop stays rolled; k-loop unrolled 2x max -> bounded liveness, no
// spills; WMMA results consumed in-register each head iteration.
// ---------------------------------------------------------------------------
template <int H, int Dm>
__device__ __forceinline__ void cfg_accum(
    const __bf16* __restrict__ Q, const __bf16* __restrict__ Kp,
    const float* __restrict__ Wp, const float* __restrict__ probs, int cfg,
    int b, int t0, int s0, int l, int hi, int mb,
    float acc0[8], float acc1[8]) {
    const __bf16* qrow  = Q  + ((size_t)(b * TN + t0 + l)) * (H * Dm) + hi * 16;
    const __bf16* krow0 = Kp + ((size_t)(b * TN + s0 + l)) * Dm + hi * 16;
    const __bf16* krow1 = krow0 + (size_t)16 * Dm;   // next s-tile, same layout
    const float*  wrow  = Wp + (size_t)(b * TN + t0 + mb) * H;

    // Per-row config probability (shared by both s-tiles of the strip).
    float pv[8];
#pragma unroll
    for (int r = 0; r < 8; ++r)
        pv[r] = probs[(size_t)(b * TN + t0 + mb + r) * 3 + cfg];

#pragma unroll 1
    for (int h = 0; h < H; ++h) {
        v8f c0 = {}, c1 = {};
        const __bf16* qh = qrow + h * Dm;
#pragma unroll 2
        for (int k = 0; k < Dm; k += 32) {
            v16bf a   = load_a16(qh + k);            // shared Q fragment
            v16bf bm0 = *(const v16bf*)(krow0 + k);  // contiguous K rows
            v16bf bm1 = *(const v16bf*)(krow1 + k);
            c0 = WMMA_BF16(a, bm0, c0);
            c1 = WMMA_BF16(a, bm1, c1);
        }
#pragma unroll
        for (int r = 0; r < 8; ++r) {
            // probs folded into head weight: w[b,t,h] * probs[b,t,cfg]
            float wp = wrow[(size_t)r * H + h] * pv[r];
            acc0[r] += fmaxf(c0[r], 0.f) * wp;       // relu(dots) * w * p
            acc1[r] += fmaxf(c1[r], 0.f) * wp;
        }
    }
}

// ---------------------------------------------------------------------------
// Fused score kernel: each wave computes a 16x32 strip (2 s-tiles), reusing
// the Q fragment across both; all three configs accumulated in-register and
// final[b,t,s] stored exactly once.
// ---------------------------------------------------------------------------
__global__ void score_kernel(
    const __bf16* __restrict__ q0, const __bf16* __restrict__ k0, const float* __restrict__ w0,
    const __bf16* __restrict__ q1, const __bf16* __restrict__ k1, const float* __restrict__ w1,
    const __bf16* __restrict__ q2, const __bf16* __restrict__ k2, const float* __restrict__ w2,
    const float* __restrict__ probs, float* __restrict__ out) {
    const int NT  = TN / 16;   // t-tiles: 128
    const int NS2 = NT / 2;    // s strip pairs: 64
    int tile = blockIdx.x * (blockDim.x >> 5) + (threadIdx.x >> 5);
    int b = tile / (NT * NS2);
    int rem = tile % (NT * NS2);
    int t0 = (rem / NS2) << 4;
    int s0 = (rem % NS2) << 5;            // 32-wide strip origin
    int lane = threadIdx.x & 31;
    int l = lane & 15, hi = lane >> 4, mb = hi << 3;

    float acc0[8] = {0.f, 0.f, 0.f, 0.f, 0.f, 0.f, 0.f, 0.f};
    float acc1[8] = {0.f, 0.f, 0.f, 0.f, 0.f, 0.f, 0.f, 0.f};

    cfg_accum<4, 32>(q0, k0, w0, probs, 0, b, t0, s0, l, hi, mb, acc0, acc1);
    cfg_accum<8, 64>(q1, k1, w1, probs, 1, b, t0, s0, l, hi, mb, acc0, acc1);
    cfg_accum<16, 128>(q2, k2, w2, probs, 2, b, t0, s0, l, hi, mb, acc0, acc1);

#pragma unroll
    for (int r = 0; r < 8; ++r) {
        size_t rowoff = ((size_t)b * TN + t0 + mb + r) * TN;
        out[rowoff + s0 + l]      = acc0[r];
        out[rowoff + s0 + 16 + l] = acc1[r];
    }
}

// ---------------------------------------------------------------------------
// Host-side orchestration
// ---------------------------------------------------------------------------
extern "C" void kernel_launch(void* const* d_in, const int* in_sizes, int n_in,
                              void* d_out, int out_size, void* d_ws, size_t ws_size,
                              hipStream_t stream) {
    (void)in_sizes; (void)n_in; (void)out_size; (void)ws_size;
    const int heads[3] = {4, 8, 16};
    const int dims[3]  = {32, 64, 128};

    const float* x      = (const float*)d_in[0];
    const float* sel_w1 = (const float*)d_in[1];
    const float* sel_b1 = (const float*)d_in[2];
    const float* sel_w2 = (const float*)d_in[3];
    const float* sel_b2 = (const float*)d_in[4];
    const float* sel_w3 = (const float*)d_in[5];
    const float* sel_b3 = (const float*)d_in[6];
    const float* qw[3] = {(const float*)d_in[7],  (const float*)d_in[10], (const float*)d_in[13]};
    const float* kw[3] = {(const float*)d_in[8],  (const float*)d_in[11], (const float*)d_in[14]};
    const float* ww[3] = {(const float*)d_in[9],  (const float*)d_in[12], (const float*)d_in[15]};
    float* out = (float*)d_out;

    // Workspace carve-out (256B aligned slabs), ~48 MB total.
    char* p = (char*)d_ws;
    auto alloc = [&](size_t bytes) -> void* {
        void* r = (void*)p;
        p += (bytes + 255) & ~(size_t)255;
        return r;
    };
    const size_t BT = (size_t)BN * TN;
    __bf16* xb   = (__bf16*)alloc(BT * DN * 2);
    __bf16* fb   = (__bf16*)alloc(BT * DN * 2);
    float*  mean = (float*)alloc((size_t)BN * DN * 4);
    float*  stdv = (float*)alloc((size_t)BN * DN * 4);
    __bf16* w1b  = (__bf16*)alloc((size_t)64 * DN * 2);
    __bf16* w2b  = (__bf16*)alloc((size_t)64 * 64 * 2);
    __bf16* h1b  = (__bf16*)alloc(BT * 64 * 2);
    __bf16* h2b  = (__bf16*)alloc(BT * 64 * 2);
    float*  prob = (float*)alloc(BT * 3 * 4);
    __bf16 *qwb[3], *kwb[3], *qb[3], *kb[3];
    float  *wv[3];
    for (int i = 0; i < 3; ++i) {
        const int hd = heads[i] * dims[i];
        qwb[i] = (__bf16*)alloc((size_t)hd * DN * 2);
        kwb[i] = (__bf16*)alloc((size_t)dims[i] * DN * 2);
        qb[i]  = (__bf16*)alloc(BT * hd * 2);
        kb[i]  = (__bf16*)alloc(BT * dims[i] * 2);
        wv[i]  = (float*)alloc(BT * heads[i] * 4);
    }

    // 1) stats + features + bf16 conversions
    stats_kernel<<<BN * DN, 256, 0, stream>>>(x, mean, stdv);
    {
        size_t tot = BT * DN;
        feats_kernel<<<(int)((tot + 255) / 256), 256, 0, stream>>>(x, mean, stdv, xb, fb);
    }
    auto conv = [&](const float* src, __bf16* dst, size_t n) {
        f32_to_bf16<<<(int)((n + 255) / 256), 256, 0, stream>>>(src, dst, n);
    };
    conv(sel_w1, w1b, (size_t)64 * DN);
    conv(sel_w2, w2b, (size_t)64 * 64);
    for (int i = 0; i < 3; ++i) {
        conv(qw[i], qwb[i], (size_t)heads[i] * dims[i] * DN);
        conv(kw[i], kwb[i], (size_t)dims[i] * DN);
    }

    // 2) WMMA GEMMs (4-wide N strips when N%64==0, else 2-wide; all N%32==0)
    auto gemm = [&](const __bf16* A, const __bf16* W, const float* bias,
                    float* oF, __bf16* oB, int M, int N, int K, int relu) {
        if (N % 64 == 0) {
            int tiles = (M >> 4) * (N >> 6);
            gemm16_bf16<4><<<(tiles + 3) / 4, 128, 0, stream>>>(A, W, bias, oF, oB, M, N, K, relu);
        } else {
            int tiles = (M >> 4) * (N >> 5);
            gemm16_bf16<2><<<(tiles + 3) / 4, 128, 0, stream>>>(A, W, bias, oF, oB, M, N, K, relu);
        }
    };
    gemm(fb,  w1b, sel_b1, nullptr, h1b, (int)BT, 64, DN, 1);   // selector L1
    gemm(h1b, w2b, sel_b2, nullptr, h2b, (int)BT, 64, 64, 1);   // selector L2
    sel3_softmax<<<(int)((BT + 255) / 256), 256, 0, stream>>>(h2b, sel_w3, sel_b3, prob, (int)BT);

    for (int i = 0; i < 3; ++i) {
        const int hd = heads[i] * dims[i];
        gemm(xb, qwb[i], nullptr, nullptr, qb[i], (int)BT, hd,      DN, 0);  // Q proj
        gemm(xb, kwb[i], nullptr, nullptr, kb[i], (int)BT, dims[i], DN, 0);  // K proj
        int nthr = (int)BT * heads[i];
        matvec_f32<<<(nthr + 255) / 256, 256, 0, stream>>>(x, ww[i], wv[i], (int)BT, heads[i], DN);
    }

    // 3) fused attention-score / blend kernel (16x32 strips)
    {
        int strips = BN * (TN / 16) * (TN / 32);       // 2*128*64 = 16384
        score_kernel<<<strips / 4, 128, 0, stream>>>(qb[0], kb[0], wv[0],
                                                     qb[1], kb[1], wv[1],
                                                     qb[2], kb[2], wv[2],
                                                     prob, out);
    }
}
